// CRF_31834297598685
// MI455X (gfx1250) — matrix operand order repaired and checked by reading, not compiled, not run on hardware
//
#include <hip/hip_runtime.h>
#include <hip/hip_bf16.h>

// CRF NLL on gfx1250 (MI455X).
// Forward algorithm in linear space: per time step,
//   acc = exp(alpha - s) @ exp(transitions)    (16x64 @ 64x64, f16 in / f32 acc)
// via 2x v_wmma_f32_16x16x32_f16 (independent, summed), then
// alpha' = s + log(acc) + e. Stability shift = reference-column g[row][0]
// (rel[col0]==0 -> acc spread <= +-10%, rel bounded by emission spread).
// Mask is monotone (t < length): gating is a register compare; rel and s live
// in registers (same lane owns the same [row][col] every step).

typedef __attribute__((ext_vector_type(16))) _Float16 v16h;
typedef __attribute__((ext_vector_type(8)))  _Float16 v8h;
typedef __attribute__((ext_vector_type(8)))  float    v8f;

#define Bsz 1024
#define Tt  512
#define Cc  53
#define CP  64
#define BOS 1
#define EOS 2
#define ROWSTRIDE (Tt * Cc)   // 27136 elements between batch rows (folds to imm offset)

#define NEG_INF (-__builtin_inff())
#define LN2f    0.6931471805599453f
#define LOG2Ef  1.4426950408889634f

static __device__ __forceinline__ v16h cat16(v8h lo, v8h hi) {
  return __builtin_shufflevector(lo, hi, 0, 1, 2, 3, 4, 5, 6, 7,
                                 8, 9, 10, 11, 12, 13, 14, 15);
}

// ---------------------------------------------------------------------------
// Gold-path score: one wave per batch row; accumulates -score into out.
// ---------------------------------------------------------------------------
__global__ __launch_bounds__(256) void crf_scores(
    const float* __restrict__ em, const int* __restrict__ tags,
    const float* __restrict__ mask, const float* __restrict__ trans,
    float* __restrict__ out) {
  const int wave = threadIdx.x >> 5;
  const int lane = threadIdx.x & 31;
  const int b = blockIdx.x * (blockDim.x >> 5) + wave;
  if (b >= Bsz) return;

  const int*   tb = tags + (size_t)b * Tt;
  const float* eb = em + (size_t)b * Tt * Cc;
  const float* mb = mask + (size_t)b * Tt;

  float acc = 0.f;
  float msum = 0.f;
  for (int t = 2 + lane; t < Tt; t += 32) {
    const float mv = mb[t];
    const int tg = tb[t];
    const int tp = tb[t - 1];
    acc += (eb[(size_t)t * Cc + tg] + trans[tp * Cc + tg]) * mv;
  }
  for (int t = lane; t < Tt; t += 32) msum += mb[t];

#pragma unroll
  for (int off = 16; off > 0; off >>= 1) {
    acc  += __shfl_xor(acc,  off, 32);
    msum += __shfl_xor(msum, off, 32);
  }

  if (lane == 0) {
    const int t1 = tb[1];
    float score = trans[BOS * Cc + t1] + eb[(size_t)1 * Cc + t1] + acc;
    const int last_idx = (int)(msum + 0.5f);   // == lengths[b] < T
    const int lt = tb[last_idx];
    score += trans[lt * Cc + EOS];
    atomicAdd(out, -score);
  }
}

// ---------------------------------------------------------------------------
// Log-partition: 16 batch rows per block, 4 waves; each wave owns one 16-col
// N-tile of the 64-col (padded) alpha state. Two independent WMMAs per step.
// ---------------------------------------------------------------------------
__global__ __launch_bounds__(128) void crf_partition(
    const float* __restrict__ em, const float* __restrict__ mask,
    const float* __restrict__ trans, float* __restrict__ out) {
  __shared__ __align__(16) _Float16 expTt[CP][CP];  // expTt[n][k]=exp(trans[k][n]), 0-pad
  __shared__ __align__(16) _Float16 expa[16][CP];   // exp(alpha - s), f16 (A operand)
  __shared__ __align__(16) float    rel[16][CP];    // alpha - s (written once, post-loop)
  __shared__ __align__(16) float    deltaArr[16];   // per-row shift g[row][0] this step
  __shared__ __align__(16) float    lenPart[16][8]; // partial mask sums
  __shared__ float sArr[16];                        // final shift s per row
  __shared__ int   lenArr[16];                      // sequence length per row

  const int tid  = threadIdx.x;
  const int wave = tid >> 5;
  const int lane = tid & 31;
  const int bbase = blockIdx.x * 16;
  const int n0 = wave * 16;
  const int nlane = n0 + (lane & 15);          // this lane's column (B/C/D layout)
  const int rowoff = (lane >> 4) << 3;         // lanes 0-15: rows r; 16-31: rows 8+r
  const int ncl = (nlane < Cc) ? nlane : (Cc - 1);      // clamped column (safe loads)
  const float colpen = (nlane < Cc) ? 0.f : NEG_INF;    // init only; loop pad g via log(0)
  const bool col0owner = (wave == 0) && ((lane & 15) == 0);  // holds g[row][0]

  // ---- one-time setup: exp(transitions)^T, per-row lengths ----
  for (int i = tid; i < CP * CP; i += blockDim.x) {
    const int n = i >> 6, k = i & 63;
    float v = 0.f;
    if (n < Cc && k < Cc)
      v = __builtin_amdgcn_exp2f(trans[k * Cc + n] * LOG2Ef);
    expTt[n][k] = (_Float16)v;
  }
  {
    const int prow = tid >> 3, pseg = tid & 7;   // 128 threads: 16 rows x 8 segs of 64
    const int mbase = (bbase + prow) * Tt + pseg * 64;
    float s = 0.f;
#pragma unroll 4
    for (int t = 0; t < 64; t++) s += mask[mbase + t];
    lenPart[prow][pseg] = s;
  }
  __syncthreads();

  if (tid < 16) {
    float s = 0.f;
#pragma unroll
    for (int j = 0; j < 8; j++) s += lenPart[tid][j];
    lenArr[tid] = (int)(s + 0.5f);               // == lengths[b]
  }

  // B operand registers (constant across all 510 steps).
  // 32x16 f16 B layout: lanes 0-15 hold K {0..7,16..23} of col N=lane; lanes 16-31 K {8..15,24..31}.
  v16h bk0, bk1;
  {
    const int N = n0 + (lane & 15);
    const int kb = (lane < 16) ? 0 : 8;
    bk0 = cat16(*(const v8h*)&expTt[N][kb],      *(const v8h*)&expTt[N][kb + 16]);
    bk1 = cat16(*(const v8h*)&expTt[N][32 + kb], *(const v8h*)&expTt[N][32 + kb + 16]);
  }
  __syncthreads();

  int lenr[8];
#pragma unroll
  for (int r = 0; r < 8; r++) lenr[r] = lenArr[r + rowoff];

  // Single running 32-bit emission offset; rows reached via constant
  // r*ROWSTRIDE folded into each instruction's immediate offset.
  int ebase = ((bbase + rowoff) * Tt + 1) * Cc + ncl;   // row rowoff, t=1

  float g[8], relq[8], sreg[8];

  // ----- init: alpha0 = trans[BOS,:] + emissions[:,1,:] -----
  {
    const float tb0 = trans[BOS * Cc + ncl];
#pragma unroll
    for (int r = 0; r < 8; r++) g[r] = tb0 + em[ebase + r * ROWSTRIDE] + colpen;
    if (col0owner) {
      const float4 w0 = {g[0], g[1], g[2], g[3]};
      const float4 w1 = {g[4], g[5], g[6], g[7]};
      *(float4*)&deltaArr[rowoff]     = w0;
      *(float4*)&deltaArr[rowoff + 4] = w1;
    }
#pragma unroll
    for (int r = 0; r < 8; r++) sreg[r] = g[r];   // s = g[row][col]; col0's is real
    __syncthreads();
    const float4 d0 = *(const float4*)&deltaArr[rowoff];
    const float4 d1 = *(const float4*)&deltaArr[rowoff + 4];
    const float dv[8] = {d0.x, d0.y, d0.z, d0.w, d1.x, d1.y, d1.z, d1.w};
#pragma unroll
    for (int r = 0; r < 8; r++) {
      relq[r] = g[r] - dv[r];
      expa[r + rowoff][nlane] = (_Float16)__builtin_amdgcn_exp2f(relq[r] * LOG2Ef);
    }
    __syncthreads();
    ebase += Cc;                                 // advance to t=2
  }

  // ----- main recursion: t = 2 .. T-1 (510 steps) -----
  for (int k = 0; k < Tt - 2; k++) {
    const int t = k + 2;

    // ---- Phase A ----
    float e[8];
#pragma unroll
    for (int r = 0; r < 8; r++) e[r] = em[ebase + r * ROWSTRIDE];

    // A operand (16x32 f16): lanes 0-15 row=lane, K {0..7,16..23};
    // lanes 16-31 row=lane-16, K {8..15,24..31}.
    const int arow = lane & 15;
    const int koff = (lane < 16) ? 0 : 8;
    v16h a0 = cat16(*(const v8h*)&expa[arow][koff],
                    *(const v8h*)&expa[arow][16 + koff]);
    v16h a1 = cat16(*(const v8h*)&expa[arow][32 + koff],
                    *(const v8h*)&expa[arow][48 + koff]);

    v8f z = {};
    v8f acc0 = __builtin_amdgcn_wmma_f32_16x16x32_f16(false, a0, false, bk0,
                                                      (short)0, z, false, false);
    v8f acc1 = __builtin_amdgcn_wmma_f32_16x16x32_f16(false, a1, false, bk1,
                                                      (short)0, z, false, false);
    v8f acc = acc0 + acc1;

    // Prefetch two steps ahead every other iteration (row advance = 212B/step).
    if (((k & 1) == 0) && (k < Tt - 4)) {
#pragma unroll
      for (int r = 0; r < 8; r++)
        __builtin_prefetch(&em[ebase + r * ROWSTRIDE + 2 * Cc], 0, 3);
    }

    // g = log(acc) + e; pad columns: acc==0 exactly -> log gives -inf.
#pragma unroll
    for (int r = 0; r < 8; r++)
      g[r] = __builtin_amdgcn_logf(acc[r]) * LN2f + e[r];

    // Column-0 lanes publish the per-row shift (two b128 stores).
    if (col0owner) {
      const float4 w0 = {g[0], g[1], g[2], g[3]};
      const float4 w1 = {g[4], g[5], g[6], g[7]};
      *(float4*)&deltaArr[rowoff]     = w0;
      *(float4*)&deltaArr[rowoff + 4] = w1;
    }
    __syncthreads();

    // ---- Phase B: branch-free gated update of register state + expa ----
    const float4 d0 = *(const float4*)&deltaArr[rowoff];
    const float4 d1 = *(const float4*)&deltaArr[rowoff + 4];
    const float dv[8] = {d0.x, d0.y, d0.z, d0.w, d1.x, d1.y, d1.z, d1.w};
    float rln[8];
#pragma unroll
    for (int r = 0; r < 8; r++) rln[r] = g[r] - dv[r];   // unconditional: keeps dv loads hoisted
#pragma unroll
    for (int r = 0; r < 8; r++) {
      const bool live = (t < lenr[r]);
      sreg[r] += live ? g[r] : 0.f;       // running shift (col0 lanes' value is real)
      const float rl = live ? rln[r] : relq[r];
      relq[r] = rl;
      expa[r + rowoff][nlane] = (_Float16)__builtin_amdgcn_exp2f(rl * LOG2Ef);
    }
    __syncthreads();

    ebase += Cc;
  }

  // ----- finalize: logZ = s + log sum_j exp(rel_j + trans[j,EOS]) -----
#pragma unroll
  for (int r = 0; r < 8; r++) rel[r + rowoff][nlane] = relq[r];
  if (col0owner) {
#pragma unroll
    for (int r = 0; r < 8; r++) sArr[r + rowoff] = sreg[r];
  }
  __syncthreads();

  if (wave == 0 && lane < 16) {
    const int row = lane;
    float sum = 0.f;
    for (int j = 0; j < Cc; j++)
      sum += __builtin_amdgcn_exp2f((rel[row][j] + trans[j * Cc + EOS]) * LOG2Ef);
    const float logZ = sArr[row] + __builtin_amdgcn_logf(sum) * LN2f;
    atomicAdd(out, logZ);
  }
}

// ---------------------------------------------------------------------------
extern "C" void kernel_launch(void* const* d_in, const int* in_sizes, int n_in,
                              void* d_out, int out_size, void* d_ws, size_t ws_size,
                              hipStream_t stream) {
  (void)in_sizes; (void)n_in; (void)out_size; (void)d_ws; (void)ws_size;
  const float* em    = (const float*)d_in[0];   // (B,T,C) f32
  const int*   tags  = (const int*)d_in[1];     // (B,T)   i32
  const float* mask  = (const float*)d_in[2];   // (B,T)   f32
  const float* trans = (const float*)d_in[3];   // (C,C)   f32
  float* out = (float*)d_out;                   // scalar f32

  hipMemsetAsync(out, 0, sizeof(float), stream);
  crf_scores<<<Bsz / 8, 256, 0, stream>>>(em, tags, mask, trans, out);
  crf_partition<<<Bsz / 16, 128, 0, stream>>>(em, mask, trans, out);
}
